// LinearTetrahedron_77721728188707
// MI455X (gfx1250) — matrix operand order repaired and checked by reading, not compiled, not run on hardware
//
#include <hip/hip_runtime.h>
#include <hip/hip_bf16.h>
#include <stdint.h>

typedef __bf16 bf16;
typedef __attribute__((ext_vector_type(16))) __bf16   v16bf;
typedef __attribute__((ext_vector_type(8)))  float    v8f;
typedef __attribute__((ext_vector_type(4)))  uint32_t u32x4;

union Frag16 { u32x4 u[2]; v16bf v; };

__device__ const int EDGE_V[6][2] = {{0,1},{0,2},{0,3},{1,2},{1,3},{2,3}};
__device__ const int FACE_V[4][3] = {{0,1,2},{0,1,3},{0,2,3},{1,2,3}};

// ---- shared memory layout (bytes) ----
constexpr int SM_XBF    = 0;                               // 16 x 512 bf16
constexpr int SM_VERTF  = SM_XBF    + 16*512*2;            // 16 x 1024 f32
constexpr int SM_VERTBF = SM_VERTF  + 16*1024*4;           // 16 x 1024 bf16
constexpr int SM_Q      = SM_VERTBF + 16*1024*2;           // 16 x 3 x 256 f32
constexpr int SM_K      = SM_Q      + 16*3*256*4;
constexpr int SM_V      = SM_K      + 16*3*256*4;
constexpr int SM_SC     = SM_V      + 16*3*256*4;          // 16 x 3 x 3 f32
constexpr int SM_TOTAL  = SM_SC     + 16*9*4;              // 262720 bytes

// A fragment: 16x32 bf16, lane(0..15)=row M, VGPR0-3 = K[k0+hi*8 .. +8),
// VGPR4-7 = K[k0+16+hi*8 .. +8)  (ISA 7.12.2, 16-bit A)
__device__ __forceinline__ v16bf load_a_frag(const bf16* base, int stride, int k0, int lane) {
  const int m = lane & 15, hi = lane >> 4;
  const bf16* p = base + m * stride + k0 + hi * 8;
  Frag16 f;
  f.u[0] = *(const u32x4*)(p);
  f.u[1] = *(const u32x4*)(p + 16);
  return f.v;
}

// B fragment: 32x16 bf16 from N-major (transposed) weights; lane n holds
// column n, VGPR0-3 = K[k0+hi*8 .. +8), VGPR4-7 = K[k0+16+hi*8 .. +8)
__device__ __forceinline__ v16bf load_b_frag(const bf16* wt, int K, int ntile, int k0, int lane) {
  const int n = lane & 15, hi = lane >> 4;
  const bf16* p = wt + (size_t)(ntile * 16 + n) * K + k0 + hi * 8;
  Frag16 f;
  f.u[0] = *(const u32x4*)(p);
  f.u[1] = *(const u32x4*)(p + 16);
  return f.v;
}

__device__ __forceinline__ v8f wmma_bf16(v16bf a, v16bf b, v8f c) {
  return __builtin_amdgcn_wmma_f32_16x16x32_bf16(false, a, false, b, (short)0, c, false, false);
}

// QKV projection tiles for one attention group: P tokens x {q,k,v} x 16 N-tiles
__device__ __forceinline__ void qkv_tiles(const bf16* s_vertbf,
                                          const bf16* wq, const bf16* wk, const bf16* wv,
                                          const float* bq, const float* bk, const float* bv,
                                          float* s_q, float* s_k, float* s_v,
                                          const int* verts, int P, int wave, int lane) {
  const int m = lane & 15, hi = lane >> 4;
  const int njobs = P * 3 * 16;
  for (int j = wave; j < njobs; j += 8) {       // wave-uniform -> EXEC all ones at WMMA
    const int p   = j / 48;
    const int mat = (j >> 4) % 3;
    const int nt  = j & 15;
    const bf16*  wt  = (mat == 0) ? wq : (mat == 1) ? wk : wv;
    const float* bp  = (mat == 0) ? bq : (mat == 1) ? bk : bv;
    float*       dst = (mat == 0) ? s_q : (mat == 1) ? s_k : s_v;
    const bf16* abase = s_vertbf + verts[p] * 256;
    const float bb = bp[nt * 16 + m];
    v8f acc;
    #pragma unroll
    for (int r = 0; r < 8; ++r) acc[r] = bb;
    #pragma unroll 4
    for (int k0 = 0; k0 < 256; k0 += 32) {
      v16bf a = load_a_frag(abase, 1024, k0, lane);
      v16bf b = load_b_frag(wt, 256, nt, k0, lane);
      acc = wmma_bf16(a, b, acc);
    }
    #pragma unroll
    for (int r = 0; r < 8; ++r) {
      const int M = r + 8 * hi;
      const int N = nt * 16 + m;
      dst[(M * 3 + p) * 256 + N] = acc[r];
    }
  }
}

// scores -> softmax -> out.mean(p) -> residual add (weight 1/3) into f32 vertices
template <int P>
__device__ __forceinline__ void attention_update(const float* s_q, const float* s_k,
                                                 const float* s_v, float* s_sc,
                                                 float* s_vertf, const int* verts, int tid) {
  __syncthreads();
  if (tid < 16 * P * P) {
    const int row = tid / (P * P);
    const int p   = (tid / P) % P;
    const int qi  = tid % P;
    const float* qp = s_q + (row * 3 + p) * 256;
    const float* kp = s_k + (row * 3 + qi) * 256;
    float s = 0.f;
    #pragma unroll 8
    for (int d = 0; d < 256; ++d) s = fmaf(qp[d], kp[d], s);
    s_sc[(row * P + p) * P + qi] = s * 0.0625f;   // 1/sqrt(256)
  }
  __syncthreads();
  {
    const int row = tid >> 4;
    const int d0  = (tid & 15) << 4;
    float cw[P];
    #pragma unroll
    for (int qi = 0; qi < P; ++qi) cw[qi] = 0.f;
    #pragma unroll
    for (int p = 0; p < P; ++p) {
      float mx = -1e30f;
      #pragma unroll
      for (int qi = 0; qi < P; ++qi) mx = fmaxf(mx, s_sc[(row * P + p) * P + qi]);
      float e[P]; float sum = 0.f;
      #pragma unroll
      for (int qi = 0; qi < P; ++qi) { e[qi] = __expf(s_sc[(row * P + p) * P + qi] - mx); sum += e[qi]; }
      const float inv = 1.f / sum;
      #pragma unroll
      for (int qi = 0; qi < P; ++qi) cw[qi] += e[qi] * inv;
    }
    const float cf = (1.f / (float)P) * (1.f / 3.f);   // mean over tokens * incidence 1/3
    #pragma unroll
    for (int qi = 0; qi < P; ++qi) cw[qi] *= cf;
    #pragma unroll 4
    for (int dd = 0; dd < 16; ++dd) {
      const int d = d0 + dd;
      float o = 0.f;
      #pragma unroll
      for (int qi = 0; qi < P; ++qi) o = fmaf(cw[qi], s_v[(row * 3 + qi) * 256 + d], o);
      #pragma unroll
      for (int vv = 0; vv < P; ++vv) s_vertf[row * 1024 + verts[vv] * 256 + d] += o;
    }
  }
  __syncthreads();
}

__global__ __launch_bounds__(256)
void tet_fused_kernel(const float* __restrict__ x,
                      const float* __restrict__ b_embed,
                      const float* __restrict__ ebq, const float* __restrict__ ebk,
                      const float* __restrict__ ebv,
                      const float* __restrict__ fbq, const float* __restrict__ fbk,
                      const float* __restrict__ fbv,
                      const float* __restrict__ b_out,
                      const bf16* __restrict__ wembT,
                      const bf16* __restrict__ eWqT, const bf16* __restrict__ eWkT,
                      const bf16* __restrict__ eWvT,
                      const bf16* __restrict__ fWqT, const bf16* __restrict__ fWkT,
                      const bf16* __restrict__ fWvT,
                      const bf16* __restrict__ woutT,
                      float* __restrict__ out) {
  extern __shared__ char smem[];
  bf16*  s_xbf    = (bf16*)(smem + SM_XBF);
  float* s_vertf  = (float*)(smem + SM_VERTF);
  bf16*  s_vertbf = (bf16*)(smem + SM_VERTBF);
  float* s_q      = (float*)(smem + SM_Q);
  float* s_k      = (float*)(smem + SM_K);
  float* s_v      = (float*)(smem + SM_V);
  float* s_sc     = (float*)(smem + SM_SC);

  const int tid  = threadIdx.x;
  const int lane = tid & 31;
  const int wave = tid >> 5;
  const int m    = lane & 15;
  const int hi   = lane >> 4;
  const size_t row0 = (size_t)blockIdx.x * 16;

  // ---- stage x tile as bf16 ----
  for (int i = tid; i < 16 * 512; i += 256)
    s_xbf[i] = (bf16)x[(row0 + (size_t)(i >> 9)) * 512 + (i & 511)];
  __syncthreads();

  // ---- embed: vertices = x @ W_embed + b_embed ----
  for (int nt = wave; nt < 64; nt += 8) {
    const float bb = b_embed[nt * 16 + m];
    v8f acc;
    #pragma unroll
    for (int r = 0; r < 8; ++r) acc[r] = bb;
    #pragma unroll 4
    for (int k0 = 0; k0 < 512; k0 += 32) {
      v16bf a = load_a_frag(s_xbf, 512, k0, lane);
      v16bf b = load_b_frag(wembT, 512, nt, k0, lane);
      acc = wmma_bf16(a, b, acc);
    }
    #pragma unroll
    for (int r = 0; r < 8; ++r) {
      const int M = r + 8 * hi;
      const int N = nt * 16 + m;
      s_vertf[M * 1024 + N]  = acc[r];
      s_vertbf[M * 1024 + N] = (bf16)acc[r];
    }
  }
  __syncthreads();

  // ---- edge attention (6 groups, P=2); all groups read pre-edge snapshot ----
  for (int g = 0; g < 6; ++g) {
    qkv_tiles(s_vertbf, eWqT + g * 65536, eWkT + g * 65536, eWvT + g * 65536,
              ebq + g * 256, ebk + g * 256, ebv + g * 256,
              s_q, s_k, s_v, EDGE_V[g], 2, wave, lane);
    attention_update<2>(s_q, s_k, s_v, s_sc, s_vertf, EDGE_V[g], tid);
  }
  for (int i = tid; i < 16 * 1024; i += 256) s_vertbf[i] = (bf16)s_vertf[i];
  __syncthreads();

  // ---- face attention (4 groups, P=3) ----
  for (int g = 0; g < 4; ++g) {
    qkv_tiles(s_vertbf, fWqT + g * 65536, fWkT + g * 65536, fWvT + g * 65536,
              fbq + g * 256, fbk + g * 256, fbv + g * 256,
              s_q, s_k, s_v, FACE_V[g], 3, wave, lane);
    attention_update<3>(s_q, s_k, s_v, s_sc, s_vertf, FACE_V[g], tid);
  }
  for (int i = tid; i < 16 * 1024; i += 256) s_vertbf[i] = (bf16)s_vertf[i];
  __syncthreads();

  // ---- output: out = vertices @ W_out + b_out ----
  for (int nt = wave; nt < 32; nt += 8) {
    const float bb = b_out[nt * 16 + m];
    v8f acc;
    #pragma unroll
    for (int r = 0; r < 8; ++r) acc[r] = bb;
    #pragma unroll 4
    for (int k0 = 0; k0 < 1024; k0 += 32) {
      v16bf a = load_a_frag(s_vertbf, 1024, k0, lane);
      v16bf b = load_b_frag(woutT, 1024, nt, k0, lane);
      acc = wmma_bf16(a, b, acc);
    }
    #pragma unroll
    for (int r = 0; r < 8; ++r) {
      const int M = r + 8 * hi;
      out[(row0 + (size_t)M) * 512 + nt * 16 + m] = acc[r];
    }
  }
}

// src: (G, K, N) f32  ->  dst: (G, N, K) bf16   (N-major rows of K for B-fragments)
__global__ void transpose_to_bf16_kernel(const float* __restrict__ src, bf16* __restrict__ dst,
                                         int G, int K, int N) {
  const long total = (long)G * K * N;
  for (long i = (long)blockIdx.x * blockDim.x + threadIdx.x; i < total;
       i += (long)gridDim.x * blockDim.x) {
    const long g = i / ((long)K * N);
    const long r = i % ((long)K * N);
    const long n = r / K;
    const long k = r % K;
    dst[i] = (bf16)src[(g * K + k) * N + n];
  }
}

extern "C" void kernel_launch(void* const* d_in, const int* in_sizes, int n_in,
                              void* d_out, int out_size, void* d_ws, size_t ws_size,
                              hipStream_t stream) {
  const float* x       = (const float*)d_in[0];
  const float* W_embed = (const float*)d_in[1];
  const float* b_embed = (const float*)d_in[2];
  const float* eWq = (const float*)d_in[3];
  const float* eWk = (const float*)d_in[4];
  const float* eWv = (const float*)d_in[5];
  const float* ebq = (const float*)d_in[6];
  const float* ebk = (const float*)d_in[7];
  const float* ebv = (const float*)d_in[8];
  const float* fWq = (const float*)d_in[9];
  const float* fWk = (const float*)d_in[10];
  const float* fWv = (const float*)d_in[11];
  const float* fbq = (const float*)d_in[12];
  const float* fbk = (const float*)d_in[13];
  const float* fbv = (const float*)d_in[14];
  const float* W_out = (const float*)d_in[15];
  const float* b_out = (const float*)d_in[16];
  float* out = (float*)d_out;

  bf16* ws = (bf16*)d_ws;
  size_t o = 0;
  bf16* wembT = ws + o; o += (size_t)1024 * 512;
  bf16* eWqT  = ws + o; o += (size_t)6 * 256 * 256;
  bf16* eWkT  = ws + o; o += (size_t)6 * 256 * 256;
  bf16* eWvT  = ws + o; o += (size_t)6 * 256 * 256;
  bf16* fWqT  = ws + o; o += (size_t)4 * 256 * 256;
  bf16* fWkT  = ws + o; o += (size_t)4 * 256 * 256;
  bf16* fWvT  = ws + o; o += (size_t)4 * 256 * 256;
  bf16* woutT = ws + o; o += (size_t)512 * 1024;

  const dim3 blk(256);
  transpose_to_bf16_kernel<<<dim3(1024), blk, 0, stream>>>(W_embed, wembT, 1, 512, 1024);
  transpose_to_bf16_kernel<<<dim3(1024), blk, 0, stream>>>(eWq, eWqT, 6, 256, 256);
  transpose_to_bf16_kernel<<<dim3(1024), blk, 0, stream>>>(eWk, eWkT, 6, 256, 256);
  transpose_to_bf16_kernel<<<dim3(1024), blk, 0, stream>>>(eWv, eWvT, 6, 256, 256);
  transpose_to_bf16_kernel<<<dim3(1024), blk, 0, stream>>>(fWq, fWqT, 4, 256, 256);
  transpose_to_bf16_kernel<<<dim3(1024), blk, 0, stream>>>(fWk, fWkT, 4, 256, 256);
  transpose_to_bf16_kernel<<<dim3(1024), blk, 0, stream>>>(fWv, fWvT, 4, 256, 256);
  transpose_to_bf16_kernel<<<dim3(1024), blk, 0, stream>>>(W_out, woutT, 1, 1024, 512);

  const int Brows = in_sizes[0] / 512;
  const int nblk  = Brows / 16;
  tet_fused_kernel<<<dim3(nblk), blk, SM_TOTAL, stream>>>(
      x, b_embed, ebq, ebk, ebv, fbq, fbk, fbv, b_out,
      wembT, eWqT, eWkT, eWvT, fWqT, fWkT, fWvT, woutT, out);
}
